// DiscreteVAE_9131100471853
// MI455X (gfx1250) — matrix operand order repaired and checked by reading, not compiled, not run on hardware
//
#include <hip/hip_runtime.h>
#include <hip/hip_bf16.h>

// ---------------------------------------------------------------------------
// DiscreteVAE forward for MI455X (gfx1250, wave32).
//  - all big GEMMs: v_wmma_f32_16x16x32_bf16, bf16 operands in global memory
//  - LDS staging via Tensor Data Mover (tensor_load_to_lds), double-buffered,
//    pipelined with s_wait_tensorcnt; plain-load fallback if builtin absent
// ---------------------------------------------------------------------------

typedef __attribute__((ext_vector_type(16))) __bf16 v16bf;
typedef __attribute__((ext_vector_type(8)))  float  v8f;

#define GEMM_BM 128
#define GEMM_BN 128
#define GEMM_BK 32
#define GEMM_LDT 40   // halves per LDS tile row: 32 data + 8 pad (80B, 16B-aligned)

#define FLAG_BIAS   1
#define FLAG_RELU   2
#define FLAG_OBF16  4

#if defined(__HIP_DEVICE_COMPILE__) && \
    __has_builtin(__builtin_amdgcn_tensor_load_to_lds) && \
    __has_builtin(__builtin_amdgcn_s_wait_tensorcnt)
#define USE_TDM 1
#else
#define USE_TDM 0
#endif

__device__ __forceinline__ unsigned short f2bf(float f) {
  union { float f; unsigned u; } x; x.f = f;
  unsigned r = x.u + 0x7FFFu + ((x.u >> 16) & 1u);   // round-to-nearest-even
  return (unsigned short)(r >> 16);
}
__device__ __forceinline__ unsigned pack2(float lo, float hi) {
  return (unsigned)f2bf(lo) | ((unsigned)f2bf(hi) << 16);
}

#if USE_TDM
typedef __attribute__((ext_vector_type(4))) unsigned int u32x4;
typedef __attribute__((ext_vector_type(4))) int          i32x4;
typedef __attribute__((ext_vector_type(8))) int          i32x8;

// Issue one TDM 2D tile load: tile = 32 (K, bf16) x 128 (rows), row stride K
// elements; LDS padded 1 DWORD-group per 16 DWORDs -> 80B row pitch.
// tensorRows < 128 makes the TDM OOB logic return zeros for missing rows.
__device__ __forceinline__ void tdm_tile_load(unsigned ldsOff,
                                              const unsigned short* g,
                                              int tensorRows, int K) {
  unsigned long long ga = (unsigned long long)(size_t)g;
  u32x4 g0;
  g0[0] = 1u;                                    // count=1, user mode
  g0[1] = ldsOff;                                // lds_addr
  g0[2] = (unsigned)ga;                          // global_addr[31:0]
  g0[3] = ((unsigned)(ga >> 32) & 0x1ffffffu) | (2u << 30);  // addr[56:32]|type=2
  i32x8 g1;
  // data_size=1(2B) | pad_enable | pad_interval=3(16 DW) | pad_amount=3(4 DW)
  g1[0] = (1 << 16) | (1 << 20) | (3 << 22) | (3 << 25);
  g1[1] = (int)(((unsigned)K & 0xffffu) << 16);                 // tensor_dim0 lo
  g1[2] = (int)(((unsigned)K >> 16) & 0xffffu) |
          (int)(((unsigned)tensorRows & 0xffffu) << 16);        // dim0 hi|dim1 lo
  g1[3] = (int)(((unsigned)tensorRows >> 16) & 0xffffu) |
          (int)(GEMM_BK << 16);                                 // dim1 hi|tile_d0
  g1[4] = GEMM_BM;                                              // tile_d1 (128)
  g1[5] = K;                                                    // dim0_stride lo
  g1[6] = 0;
  g1[7] = 0;
  i32x4 gz; gz[0] = 0; gz[1] = 0; gz[2] = 0; gz[3] = 0;
#if defined(__clang_major__) && __clang_major__ >= 23
  i32x8 g4; for (int i = 0; i < 8; ++i) g4[i] = 0;
  __builtin_amdgcn_tensor_load_to_lds(g0, g1, gz, gz, g4, 0);
#else
  __builtin_amdgcn_tensor_load_to_lds(g0, g1, gz, gz, 0);
#endif
}
#endif // USE_TDM

// ---------------------------------------------------------------------------
// GEMM: C[M,N] = act( scale * (A[M,K]bf16 @ W[N,K]bf16^T) + bias )
// M%128==0, K%32==0; N may be any multiple of 64 (edge via OOB-zero / guards).
// ---------------------------------------------------------------------------
__global__ __launch_bounds__(256)
void gemm_bf16_wmma(const unsigned short* __restrict__ A,
                    const unsigned short* __restrict__ W,
                    const float* __restrict__ bias, void* __restrict__ Cout,
                    int M, int N, int K, int flags, float scale) {
  __shared__ __align__(16) unsigned short sA[2][GEMM_BM * GEMM_LDT];
  __shared__ __align__(16) unsigned short sB[2][GEMM_BN * GEMM_LDT];

  const int tid  = threadIdx.x;
  const int lane = tid & 31;
  const int wave = tid >> 5;
  const int wm   = wave >> 1;    // 0..3 : 32-row strip
  const int wn   = wave & 1;     // 0..1 : 64-col strip
  const int bm   = blockIdx.y * GEMM_BM;
  const int bn   = blockIdx.x * GEMM_BN;
  const int nT   = K / GEMM_BK;

  v8f acc[2][4];
#pragma unroll
  for (int i = 0; i < 2; ++i)
#pragma unroll
    for (int j = 0; j < 4; ++j)
#pragma unroll
      for (int r = 0; r < 8; ++r) acc[i][j][r] = 0.0f;

#if USE_TDM
  unsigned offA[2], offB[2];
  offA[0] = (unsigned)(size_t)&sA[0][0];
  offA[1] = (unsigned)(size_t)&sA[1][0];
  offB[0] = (unsigned)(size_t)&sB[0][0];
  offB[1] = (unsigned)(size_t)&sB[1][0];
  if (tid < 32) {  // wave 0 drives the Tensor Data Mover
    tdm_tile_load(offA[0], A + (size_t)bm * K, GEMM_BM, K);
    tdm_tile_load(offB[0], W + (size_t)bn * K, N - bn, K);
  }
#endif

  for (int t = 0; t < nT; ++t) {
#if USE_TDM
    const int cb = t & 1;
    if (tid < 32) {
      if (t + 1 < nT) {
        const size_t kofs = (size_t)(t + 1) * GEMM_BK;
        tdm_tile_load(offA[(t + 1) & 1], A + (size_t)bm * K + kofs, GEMM_BM, K);
        tdm_tile_load(offB[(t + 1) & 1], W + (size_t)bn * K + kofs, N - bn, K);
        __builtin_amdgcn_s_wait_tensorcnt(2);  // tile t done, t+1 in flight
      } else {
        __builtin_amdgcn_s_wait_tensorcnt(0);
      }
    }
    __syncthreads();
#else
    const int cb = 0;
    const int k0 = t * GEMM_BK;
    {  // stage A: 2 threads/row, 16 halves each
      const int r  = tid >> 1;
      const int ch = (tid & 1) * 16;
      if (k0 + GEMM_BK < K)
        __builtin_prefetch(A + (size_t)(bm + r) * K + k0 + GEMM_BK + ch, 0, 3);
      const uint4* s4 = (const uint4*)(A + (size_t)(bm + r) * K + k0 + ch);
      uint4* d = (uint4*)&sA[0][r * GEMM_LDT + ch];
      d[0] = s4[0]; d[1] = s4[1];
    }
    {  // stage W with N-edge guard
      const int r  = tid >> 1;
      const int ch = (tid & 1) * 16;
      uint4 v0, v1;
      v0.x = v0.y = v0.z = v0.w = 0;
      v1 = v0;
      if (bn + r < N) {
        if (k0 + GEMM_BK < K)
          __builtin_prefetch(W + (size_t)(bn + r) * K + k0 + GEMM_BK + ch, 0, 3);
        const uint4* s4 = (const uint4*)(W + (size_t)(bn + r) * K + k0 + ch);
        v0 = s4[0]; v1 = s4[1];
      }
      uint4* d = (uint4*)&sB[0][r * GEMM_LDT + ch];
      d[0] = v0; d[1] = v1;
    }
    __syncthreads();
#endif

    // --- fragments: contiguous-K layout, same order for A and B ---
    union Frag { v16bf v; uint4 u[2]; };
    Frag fa[2], fb[4];
    // A (16x32 MxK): lane row M=lane%16; 16 contiguous K (0..15 / 16..31)
    const int ar0   = wm * 32 + (lane & 15);
    const int kofsA = (lane < 16) ? 0 : 16;
#pragma unroll
    for (int mi = 0; mi < 2; ++mi) {
      const unsigned short* p = &sA[cb][(ar0 + mi * 16) * GEMM_LDT + kofsA];
      fa[mi].u[0] = *(const uint4*)(p);
      fa[mi].u[1] = *(const uint4*)(p + 8);
    }
    // B (32x16 KxN): lane col N=lane%16; 16 contiguous K (0..15 / 16..31)
    const int br0   = wn * 64 + (lane & 15);
    const int kofsB = (lane < 16) ? 0 : 16;
#pragma unroll
    for (int ni = 0; ni < 4; ++ni) {
      const unsigned short* p = &sB[cb][(br0 + ni * 16) * GEMM_LDT + kofsB];
      fb[ni].u[0] = *(const uint4*)(p);
      fb[ni].u[1] = *(const uint4*)(p + 8);
    }

#pragma unroll
    for (int mi = 0; mi < 2; ++mi)
#pragma unroll
      for (int ni = 0; ni < 4; ++ni)
        acc[mi][ni] = __builtin_amdgcn_wmma_f32_16x16x32_bf16(
            false, fa[mi].v, false, fb[ni].v, (short)0, acc[mi][ni],
            false, false);
    __syncthreads();
  }

  // Epilogue. C 16x16 f32: VGPR r, lanes0-15: M=r,N=lane; lanes16-31: M=r+8,N-16
  const int crow0 = bm + wm * 32 + ((lane < 16) ? 0 : 8);
  const int ccol0 = bn + wn * 64 + (lane & 15);
  float* Cf = (float*)Cout;
  unsigned short* Ch = (unsigned short*)Cout;
#pragma unroll
  for (int mi = 0; mi < 2; ++mi)
#pragma unroll
    for (int ni = 0; ni < 4; ++ni) {
      const int col = ccol0 + ni * 16;
      if (col >= N) continue;
      const float bv = (flags & FLAG_BIAS) ? bias[col] : 0.0f;
#pragma unroll
      for (int r = 0; r < 8; ++r) {
        const int row = crow0 + mi * 16 + r;
        float v = acc[mi][ni][r] * scale + bv;
        if (flags & FLAG_RELU) v = fmaxf(v, 0.0f);
        if (flags & FLAG_OBF16) Ch[(size_t)row * N + col] = f2bf(v);
        else                    Cf[(size_t)row * N + col] = v;
      }
    }
}

// ---------------------------------------------------------------------------
// Elementwise / small kernels
// ---------------------------------------------------------------------------
__global__ void cvt_bf16_k(const float* __restrict__ x,
                           unsigned short* __restrict__ y, int n) {
  int i = blockIdx.x * blockDim.x + threadIdx.x;
  if (i < n) y[i] = f2bf(x[i]);
}

__global__ void rmsnorm_k(const float* __restrict__ x, const float* __restrict__ w,
                          unsigned short* __restrict__ y, int d) {
  __shared__ float red[8];
  __shared__ float s_scale;
  const int row = blockIdx.x;
  const float* xr = x + (size_t)row * d;
  float ss = 0.0f;
  for (int i = threadIdx.x; i < d; i += blockDim.x) { float v = xr[i]; ss += v * v; }
  for (int o = 16; o > 0; o >>= 1) ss += __shfl_xor(ss, o, 32);
  if ((threadIdx.x & 31) == 0) red[threadIdx.x >> 5] = ss;
  __syncthreads();
  if (threadIdx.x == 0) {
    float t = 0.0f;
    for (int i = 0; i < 8; ++i) t += red[i];
    s_scale = rsqrtf(t / (float)d + 1e-6f);
  }
  __syncthreads();
  const float sc = s_scale;
  unsigned short* yr = y + (size_t)row * d;
  for (int i = threadIdx.x; i < d; i += blockDim.x) yr[i] = f2bf(xr[i] * sc * w[i]);
}

// RoPE fp32 -> packed bf16 pairs
__global__ void rope_bf16_k(const float* __restrict__ in, unsigned* __restrict__ out,
                            int rows, int d, int posmod) {
  const int pairs = d / 2;
  int idx = blockIdx.x * blockDim.x + threadIdx.x;
  if (idx >= rows * pairs) return;
  int row = idx / pairs, j = idx % pairs;
  float pos   = (float)(row % posmod);
  float theta = __powf(10000.0f, -2.0f * (float)j / (float)d);
  float s, c;
  __sincosf(pos * theta, &s, &c);
  const float* p = in + (size_t)row * d + 2 * j;
  float x1 = p[0], x2 = p[1];
  out[(size_t)row * pairs + j] = pack2(x1 * c - x2 * s, x1 * s + x2 * c);
}

__global__ void softmax_gumbel_k(const float* __restrict__ la, const float* __restrict__ g,
                                 const float* __restrict__ tau, float* __restrict__ z,
                                 unsigned short* __restrict__ zh, int cols) {
  __shared__ float red[8];
  __shared__ float s_b;
  const int row = blockIdx.x;
  const float* a  = la + (size_t)row * cols;
  const float* gr = g  + (size_t)row * cols;
  float* zr = z + (size_t)row * cols;
  unsigned short* zhr = zh + (size_t)row * cols;
  const float inv_tau = 1.0f / tau[0];

  float mx = -3.4e38f;
  for (int i = threadIdx.x; i < cols; i += blockDim.x)
    mx = fmaxf(mx, (a[i] + gr[i]) * inv_tau);
  for (int o = 16; o > 0; o >>= 1) mx = fmaxf(mx, __shfl_xor(mx, o, 32));
  if ((threadIdx.x & 31) == 0) red[threadIdx.x >> 5] = mx;
  __syncthreads();
  if (threadIdx.x == 0) {
    float m = red[0];
    for (int i = 1; i < 8; ++i) m = fmaxf(m, red[i]);
    s_b = m;
  }
  __syncthreads();
  const float m = s_b;
  __syncthreads();

  float sum = 0.0f;
  for (int i = threadIdx.x; i < cols; i += blockDim.x) {
    float e = __expf((a[i] + gr[i]) * inv_tau - m);
    zr[i] = e;
    sum += e;
  }
  for (int o = 16; o > 0; o >>= 1) sum += __shfl_xor(sum, o, 32);
  if ((threadIdx.x & 31) == 0) red[threadIdx.x >> 5] = sum;
  __syncthreads();
  if (threadIdx.x == 0) {
    float t = 0.0f;
    for (int i = 0; i < 8; ++i) t += red[i];
    s_b = 1.0f / t;
  }
  __syncthreads();
  const float inv = s_b;
  for (int i = threadIdx.x; i < cols; i += blockDim.x) {
    float v = zr[i] * inv;
    zr[i]  = v;
    zhr[i] = f2bf(v);
  }
}

__global__ void silu_gate_bf16_k(const float* __restrict__ t1, const float* __restrict__ t3,
                                 unsigned short* __restrict__ out, int n) {
  int i = blockIdx.x * blockDim.x + threadIdx.x;
  if (i >= n) return;
  float a = t1[i];
  out[i] = f2bf((a / (1.0f + __expf(-a))) * t3[i]);
}

__global__ void add_dual_k(const float* __restrict__ a, const float* __restrict__ b,
                           float* __restrict__ of, unsigned short* __restrict__ oh, int n) {
  int i = blockIdx.x * blockDim.x + threadIdx.x;
  if (i >= n) return;
  float s = a[i] + b[i];
  of[i] = s;
  oh[i] = f2bf(s);
}

__global__ void transpose_bf16_k(const float* __restrict__ in,
                                 unsigned short* __restrict__ out, int rows, int cols) {
  int i = blockIdx.x * blockDim.x + threadIdx.x;
  if (i >= rows * cols) return;
  int r = i / cols, c = i % cols;
  out[(size_t)c * rows + r] = f2bf(in[i]);
}

// conv 3x3, stride 2, pad 1, bias + relu. w: (OC,IC,3,3). out fp32 or bf16.
__global__ void conv3x3_s2_relu(const float* __restrict__ x, const float* __restrict__ w,
                                const float* __restrict__ b, void* __restrict__ y,
                                int Bn, int IC, int IH, int IW, int OC, int OH, int OW,
                                int out_bf16) {
  int idx = blockIdx.x * blockDim.x + threadIdx.x;
  int total = Bn * OC * OH * OW;
  if (idx >= total) return;
  int ox = idx % OW; int t = idx / OW;
  int oy = t % OH;   t /= OH;
  int oc = t % OC;   int n = t / OC;
  float acc = b[oc];
  for (int ic = 0; ic < IC; ++ic) {
    const float* xp = x + (size_t)(n * IC + ic) * IH * IW;
    const float* wp = w + (size_t)(oc * IC + ic) * 9;
    for (int ky = 0; ky < 3; ++ky) {
      int iy = oy * 2 - 1 + ky;
      if (iy < 0 || iy >= IH) continue;
      for (int kx = 0; kx < 3; ++kx) {
        int ix = ox * 2 - 1 + kx;
        if (ix < 0 || ix >= IW) continue;
        acc += xp[iy * IW + ix] * wp[ky * 3 + kx];
      }
    }
  }
  acc = fmaxf(acc, 0.0f);
  if (out_bf16) ((unsigned short*)y)[idx] = f2bf(acc);
  else          ((float*)y)[idx] = acc;
}

// transposed conv 3x3, stride 2. w: (IC,OC,3,3). act: 0=relu, 1=sigmoid.
__global__ void deconv3x3_s2(const float* __restrict__ x, const float* __restrict__ w,
                             const float* __restrict__ b, float* __restrict__ y,
                             int Bn, int IC, int IH, int IW, int OC, int OH, int OW,
                             int act) {
  int idx = blockIdx.x * blockDim.x + threadIdx.x;
  int total = Bn * OC * OH * OW;
  if (idx >= total) return;
  int ox = idx % OW; int t = idx / OW;
  int oy = t % OH;   t /= OH;
  int oc = t % OC;   int n = t / OC;
  float acc = b[oc];
  for (int ic = 0; ic < IC; ++ic) {
    const float* xp = x + (size_t)(n * IC + ic) * IH * IW;
    const float* wp = w + (size_t)(ic * OC + oc) * 9;
    for (int ky = 0; ky < 3; ++ky) {
      int ty = oy + 1 - ky;
      if (ty < 0 || (ty & 1)) continue;
      int iy = ty >> 1;
      if (iy >= IH) continue;
      for (int kx = 0; kx < 3; ++kx) {
        int tx = ox + 1 - kx;
        if (tx < 0 || (tx & 1)) continue;
        int ix = tx >> 1;
        if (ix >= IW) continue;
        acc += xp[iy * IW + ix] * wp[ky * 3 + kx];
      }
    }
  }
  y[idx] = act ? (1.0f / (1.0f + __expf(-acc))) : fmaxf(acc, 0.0f);
}

// ---------------------------------------------------------------------------
// Host-side orchestration
// ---------------------------------------------------------------------------
static inline dim3 gemm_grid(int M, int N) {
  return dim3((N + GEMM_BN - 1) / GEMM_BN, M / GEMM_BM);
}
static inline int ceil_div(long long n, int t) { return (int)((n + t - 1) / t); }

extern "C" void kernel_launch(void* const* d_in, const int* in_sizes, int n_in,
                              void* d_out, int out_size, void* d_ws, size_t ws_size,
                              hipStream_t stream) {
  (void)in_sizes; (void)n_in; (void)out_size; (void)ws_size;
  const int Bx = 512, D = 512, Kc = 1024, Nn = 64, HID = 1024, FF = 2048;
  const float inv_sqrt_d = 0.044194173824159216f; // 1/sqrt(512)

  const float* x        = (const float*)d_in[0];
  const float* tau      = (const float*)d_in[1];
  const float* gumbel   = (const float*)d_in[2];
  const float* e_c1w    = (const float*)d_in[3];
  const float* e_c1b    = (const float*)d_in[4];
  const float* e_c2w    = (const float*)d_in[5];
  const float* e_c2b    = (const float*)d_in[6];
  const float* e_f1w    = (const float*)d_in[7];
  const float* e_f1b    = (const float*)d_in[8];
  const float* e_f2w    = (const float*)d_in[9];
  const float* e_f2b    = (const float*)d_in[10];
  const float* codebook = (const float*)d_in[11];
  const float* nctx_w   = (const float*)d_in[12];
  const float* nq_w     = (const float*)d_in[13];
  const float* nff_w    = (const float*)d_in[14];
  const float* qp_w     = (const float*)d_in[15];
  const float* kp_w     = (const float*)d_in[16];
  const float* vp_w     = (const float*)d_in[17];
  const float* vp_b     = (const float*)d_in[18];
  const float* cp_w     = (const float*)d_in[19];
  const float* ff_w1    = (const float*)d_in[20];
  const float* ff_w3    = (const float*)d_in[21];
  const float* ff_w2    = (const float*)d_in[22];
  const float* d_f1w    = (const float*)d_in[23];
  const float* d_f1b    = (const float*)d_in[24];
  const float* d_f2w    = (const float*)d_in[25];
  const float* d_f2b    = (const float*)d_in[26];
  const float* d_dc1w   = (const float*)d_in[27];
  const float* d_dc1b   = (const float*)d_in[28];
  const float* d_dc2w   = (const float*)d_in[29];
  const float* d_dc2b   = (const float*)d_in[30];

  float* out = (float*)d_out;
  float* o_recon = out;                             // 512*784
  float* o_la    = out + 401408;                    // 512*64*1024
  float* o_z     = o_la  + (size_t)33554432;
  float* o_lat   = o_z   + (size_t)33554432;        // 512*64*512
  float* o_quant = o_lat + (size_t)16777216;

  // ---- workspace carve-up: fp32 pool then bf16 pool ----
  float* fp = (float*)d_ws;
  float* f_h1  = fp; fp += 3211264;    // conv1 out / deconv1 out
  float* f_B   = fp; fp += 16777216;   // q fp32 -> attn_out fp32
  float* f_t1  = fp; fp += 16777216;   // FFN chunk t1
  float* f_t3  = fp; fp += 16777216;   // FFN chunk t3
  float* f_ff  = fp; fp += 16777216;   // FFN output (full)
  float* f_k   = fp; fp += 524288;     // k fp32 (pre-rope)
  float* f_v   = fp; fp += 524288;     // v fp32
  float* f_dec = fp; fp += 1605632;    // d_f2 out fp32

  unsigned short* hp = (unsigned short*)fp;
  unsigned short* h_wef1 = hp; hp += 3211264;
  unsigned short* h_wef2 = hp; hp += 33554432;
  unsigned short* h_wqp  = hp; hp += 262144;
  unsigned short* h_wkp  = hp; hp += 262144;
  unsigned short* h_wvp  = hp; hp += 262144;
  unsigned short* h_wcp  = hp; hp += 262144;
  unsigned short* h_wff1 = hp; hp += 1048576;
  unsigned short* h_wff3 = hp; hp += 1048576;
  unsigned short* h_wff2 = hp; hp += 1048576;
  unsigned short* h_wdf1 = hp; hp += 33554432;
  unsigned short* h_wdf2 = hp; hp += 3211264;
  unsigned short* h_h2   = hp; hp += 1605632;   // conv2 out
  unsigned short* h_enc  = hp; hp += 524288;    // e_f1 out / d_f1 out
  unsigned short* h_nctx = hp; hp += 524288;
  unsigned short* h_X    = hp; hp += 16777216;  // nq -> attn -> hf
  unsigned short* h_Y    = hp; hp += 16777216;  // q -> gate(chunk) -> quantized
  unsigned short* h_z    = hp; hp += 33554432;  // z bf16
  unsigned short* h_k    = hp; hp += 524288;    // k bf16 (post-rope)
  unsigned short* h_vT   = hp; hp += 524288;    // v^T bf16

  const int T = 256;
  const int BN_tok = Bx * Nn; // 32768 token rows

  // ---- convert weights to bf16 once ----
  cvt_bf16_k<<<ceil_div(3211264, T), T, 0, stream>>>(e_f1w, h_wef1, 3211264);
  cvt_bf16_k<<<ceil_div(33554432, T), T, 0, stream>>>(e_f2w, h_wef2, 33554432);
  cvt_bf16_k<<<ceil_div(262144, T), T, 0, stream>>>(qp_w, h_wqp, 262144);
  cvt_bf16_k<<<ceil_div(262144, T), T, 0, stream>>>(kp_w, h_wkp, 262144);
  cvt_bf16_k<<<ceil_div(262144, T), T, 0, stream>>>(vp_w, h_wvp, 262144);
  cvt_bf16_k<<<ceil_div(262144, T), T, 0, stream>>>(cp_w, h_wcp, 262144);
  cvt_bf16_k<<<ceil_div(1048576, T), T, 0, stream>>>(ff_w1, h_wff1, 1048576);
  cvt_bf16_k<<<ceil_div(1048576, T), T, 0, stream>>>(ff_w3, h_wff3, 1048576);
  cvt_bf16_k<<<ceil_div(1048576, T), T, 0, stream>>>(ff_w2, h_wff2, 1048576);
  cvt_bf16_k<<<ceil_div(33554432, T), T, 0, stream>>>(d_f1w, h_wdf1, 33554432);
  cvt_bf16_k<<<ceil_div(3211264, T), T, 0, stream>>>(d_f2w, h_wdf2, 3211264);

  // ---- Encoder ----
  conv3x3_s2_relu<<<ceil_div(3211264, T), T, 0, stream>>>(
      x, e_c1w, e_c1b, f_h1, Bx, 1, 28, 28, 32, 14, 14, 0);
  conv3x3_s2_relu<<<ceil_div(1605632, T), T, 0, stream>>>(
      f_h1, e_c2w, e_c2b, h_h2, Bx, 32, 14, 14, 64, 7, 7, 1);
  gemm_bf16_wmma<<<gemm_grid(512, HID), T, 0, stream>>>(
      h_h2, h_wef1, e_f1b, h_enc, 512, HID, 3136, FLAG_BIAS | FLAG_RELU | FLAG_OBF16, 1.0f);
  gemm_bf16_wmma<<<gemm_grid(512, Nn * D), T, 0, stream>>>(
      h_enc, h_wef2, e_f2b, o_lat, 512, Nn * D, HID, FLAG_BIAS, 1.0f);   // latent fp32

  // ---- K/V from codebook ----
  rmsnorm_k<<<Kc, T, 0, stream>>>(codebook, nctx_w, h_nctx, D);
  gemm_bf16_wmma<<<gemm_grid(Kc, D), T, 0, stream>>>(
      h_nctx, h_wkp, nullptr, f_k, Kc, D, D, 0, 1.0f);
  gemm_bf16_wmma<<<gemm_grid(Kc, D), T, 0, stream>>>(
      h_nctx, h_wvp, vp_b, f_v, Kc, D, D, FLAG_BIAS, 1.0f);
  rope_bf16_k<<<ceil_div((long long)Kc * (D / 2), T), T, 0, stream>>>(
      f_k, (unsigned*)h_k, Kc, D, Kc);                                   // pos = row
  transpose_bf16_k<<<ceil_div((long long)Kc * D, T), T, 0, stream>>>(f_v, h_vT, Kc, D);

  // ---- Q ----
  rmsnorm_k<<<BN_tok, T, 0, stream>>>(o_lat, nq_w, h_X, D);              // nq bf16
  gemm_bf16_wmma<<<gemm_grid(BN_tok, D), T, 0, stream>>>(
      h_X, h_wqp, nullptr, f_B, BN_tok, D, D, 0, 1.0f);                  // q fp32
  rope_bf16_k<<<ceil_div((long long)BN_tok * (D / 2), T), T, 0, stream>>>(
      f_B, (unsigned*)h_Y, BN_tok, D, Nn);                               // pos = n%64

  // ---- attention logits + gumbel softmax ----
  gemm_bf16_wmma<<<gemm_grid(BN_tok, Kc), T, 0, stream>>>(
      h_Y, h_k, nullptr, o_la, BN_tok, Kc, D, 0, inv_sqrt_d);            // log_alpha
  softmax_gumbel_k<<<BN_tok, T, 0, stream>>>(o_la, gumbel, tau, o_z, h_z, Kc);

  // ---- attn = (z @ v) @ cp^T ----
  gemm_bf16_wmma<<<gemm_grid(BN_tok, D), T, 0, stream>>>(
      h_z, h_vT, nullptr, h_X, BN_tok, D, Kc, FLAG_OBF16, 1.0f);         // attn bf16
  gemm_bf16_wmma<<<gemm_grid(BN_tok, D), T, 0, stream>>>(
      h_X, h_wcp, nullptr, f_B, BN_tok, D, D, 0, 1.0f);                  // attn_out fp32

  // ---- gated FFN (chunked over M) ----
  rmsnorm_k<<<BN_tok, T, 0, stream>>>(f_B, nff_w, h_X, D);               // hf bf16
  const int MCH = 8192;
  for (int m0 = 0; m0 < BN_tok; m0 += MCH) {
    const unsigned short* hfc = h_X + (size_t)m0 * D;
    gemm_bf16_wmma<<<gemm_grid(MCH, FF), T, 0, stream>>>(
        hfc, h_wff1, nullptr, f_t1, MCH, FF, D, 0, 1.0f);
    gemm_bf16_wmma<<<gemm_grid(MCH, FF), T, 0, stream>>>(
        hfc, h_wff3, nullptr, f_t3, MCH, FF, D, 0, 1.0f);
    silu_gate_bf16_k<<<ceil_div((long long)MCH * FF, T), T, 0, stream>>>(
        f_t1, f_t3, h_Y, MCH * FF);
    gemm_bf16_wmma<<<gemm_grid(MCH, D), T, 0, stream>>>(
        h_Y, h_wff2, nullptr, f_ff + (size_t)m0 * D, MCH, D, FF, 0, 1.0f);
  }
  add_dual_k<<<ceil_div((long long)BN_tok * D, T), T, 0, stream>>>(
      f_B, f_ff, o_quant, h_Y, BN_tok * D);                              // quantized

  // ---- Decoder ----
  gemm_bf16_wmma<<<gemm_grid(512, HID), T, 0, stream>>>(
      h_Y, h_wdf1, d_f1b, h_enc, 512, HID, Nn * D, FLAG_BIAS | FLAG_RELU | FLAG_OBF16, 1.0f);
  gemm_bf16_wmma<<<gemm_grid(512, 3136), T, 0, stream>>>(
      h_enc, h_wdf2, d_f2b, f_dec, 512, 3136, HID, FLAG_BIAS | FLAG_RELU, 1.0f);
  deconv3x3_s2<<<ceil_div(3211264, T), T, 0, stream>>>(
      f_dec, d_dc1w, d_dc1b, f_h1, Bx, 64, 7, 7, 32, 14, 14, 0);
  deconv3x3_s2<<<ceil_div(401408, T), T, 0, stream>>>(
      f_h1, d_dc2w, d_dc2b, o_recon, Bx, 32, 14, 14, 1, 28, 28, 1);
}
